// KGAwareGNN_76811195121824
// MI455X (gfx1250) — compile-verified, matrix operand order checked by reading, not run on hardware
//
#include <hip/hip_runtime.h>
#include <hip/hip_bf16.h>

#define B_  4
#define N_  256
#define D_  256
#define KG_ 128
#define L_  2
#define APAD 8                     // bf16 row padding: (128+8)*2B = 272B = 68 dwords
#define ALD (KG_ + APAD)

typedef __attribute__((ext_vector_type(16))) __bf16 v16bf;
typedef __attribute__((ext_vector_type(8)))  float  v8f;

// ---------------------------------------------------------------------------
// WMMA fragment loaders (layouts per CDNA5 ISA 7.12.2, wave32)
// A (16x32 bf16, MxK): elem e -> k = (e>>3)*16 + (lane>=16)*8 + (e&7), row=lane&15
// ---------------------------------------------------------------------------
__device__ __forceinline__ v16bf load_frag_A(const float* A, int ld,
                                             int row0, int k0, int lane) {
    int l16 = lane & 15, hi = lane >> 4;
    const float* p = A + (size_t)(row0 + l16) * ld + k0 + hi * 8;
    float4 x0 = *(const float4*)(p + 0);
    float4 x1 = *(const float4*)(p + 4);
    float4 x2 = *(const float4*)(p + 16);
    float4 x3 = *(const float4*)(p + 20);
    v16bf f;
    f[0]=(__bf16)x0.x; f[1]=(__bf16)x0.y; f[2]=(__bf16)x0.z; f[3]=(__bf16)x0.w;
    f[4]=(__bf16)x1.x; f[5]=(__bf16)x1.y; f[6]=(__bf16)x1.z; f[7]=(__bf16)x1.w;
    f[8]=(__bf16)x2.x; f[9]=(__bf16)x2.y; f[10]=(__bf16)x2.z; f[11]=(__bf16)x2.w;
    f[12]=(__bf16)x3.x; f[13]=(__bf16)x3.y; f[14]=(__bf16)x3.z; f[15]=(__bf16)x3.w;
    return f;
}

// B (32x16 bf16, KxN), row-major source, lane -> column (lane&15)
__device__ __forceinline__ v16bf load_frag_B(const float* Bm, int ld,
                                             int k0, int col0, int lane) {
    int l16 = lane & 15, hi = lane >> 4;
    const float* p = Bm + (size_t)(k0 + hi * 8) * ld + col0 + l16;
    v16bf f;
#pragma unroll
    for (int e = 0; e < 8; ++e) f[e]     = (__bf16)p[(size_t)e * ld];
#pragma unroll
    for (int e = 0; e < 8; ++e) f[8 + e] = (__bf16)p[(size_t)(16 + e) * ld];
    return f;
}

// A fragment from an LDS-staged bf16 tile [16][ALD]
__device__ __forceinline__ v16bf frag_A_lds(const __bf16* tile, int kc, int lane) {
    int l16 = lane & 15, hi = lane >> 4;
    const __bf16* p = tile + l16 * ALD + kc * 32 + hi * 8;
    uint4 lo  = *(const uint4*)(p);        // k: base+0..7   (ds_load_b128)
    uint4 hi4 = *(const uint4*)(p + 16);   // k: base+16..23 (ds_load_b128)
    v16bf f;
    *((uint4*)&f)     = lo;
    *((uint4*)&f + 1) = hi4;
    return f;
}

// Coalesced global->LDS staging of one 16x128 f32 tile as bf16 (256 threads)
__device__ __forceinline__ void stage_tile(const float* __restrict__ Abase,
                                           int jt, __bf16* dst, int t) {
    int flat = t * 8;                 // 2048 elements / 256 threads
    int row  = flat >> 7;             // /128
    int kb   = flat & 127;
    const float* p = Abase + (size_t)(jt * 16 + row) * KG_ + kb;
    float4 v0 = *(const float4*)(p);
    float4 v1 = *(const float4*)(p + 4);
    union { __bf16 h[8]; uint4 u; } pk;
    pk.h[0]=(__bf16)v0.x; pk.h[1]=(__bf16)v0.y; pk.h[2]=(__bf16)v0.z; pk.h[3]=(__bf16)v0.w;
    pk.h[4]=(__bf16)v1.x; pk.h[5]=(__bf16)v1.y; pk.h[6]=(__bf16)v1.z; pk.h[7]=(__bf16)v1.w;
    *(uint4*)(dst + row * ALD + kb) = pk.u;   // 16B-aligned ds_store_b128
}

// ---------------------------------------------------------------------------
// Generic GEMM: C(MxN) = A(MxK) @ B(KxN), fp32 in/out, bf16 WMMA compute.
// ---------------------------------------------------------------------------
__global__ __launch_bounds__(256)
void gemm_wmma(const float* __restrict__ A, const float* __restrict__ Bm,
               float* __restrict__ C, int M, int N, int K) {
    int lane = threadIdx.x & 31;
    int wave = threadIdx.x >> 5;
    int tilesN = N >> 4;
    int tiles  = (M >> 4) * tilesN;
    int tile   = blockIdx.x * 8 + wave;
    if (tile >= tiles) return;
    int mt = tile / tilesN, nt = tile % tilesN;

    v8f acc = {};
    for (int k0 = 0; k0 < K; k0 += 32) {
        v16bf af = load_frag_A(A, K, mt * 16, k0, lane);
        v16bf bf = load_frag_B(Bm, N, k0, nt * 16, lane);
        acc = __builtin_amdgcn_wmma_f32_16x16x32_bf16(
            false, af, false, bf, (short)0, acc, false, false);
    }
    int l16 = lane & 15, hi = lane >> 4;
    int col = nt * 16 + l16;
#pragma unroll
    for (int r = 0; r < 8; ++r)
        C[(size_t)(mt * 16 + hi * 8 + r) * N + col] = acc[r];
}

// cbias[l][d] = b_kg @ Wc[l] + b1[l]
__global__ void cbias_kernel(const float* __restrict__ b_kg,
                             const float* __restrict__ W1,
                             const float* __restrict__ b1,
                             float* __restrict__ cbias) {
    int l = blockIdx.x, d = threadIdx.x;
    const float* Wc = W1 + ((size_t)l * 3 * D_ + 2 * D_) * D_;
    float s = b1[l * D_ + d];
    for (int k = 0; k < D_; ++k) s += b_kg[k] * Wc[(size_t)k * D_ + d];
    cbias[l * D_ + d] = s;
}

// ---------------------------------------------------------------------------
// Fused message kernel: one block per (b,i).
//   s[b,i,:] = sum_j adj[b,i,j] * relu(a[b,i,:] + bfeat[b,j,:]
//                                      + rel_emb[b,i,j,:]@WcEff + cbias)
// A tiles staged in LDS once per block (double-buffered), B frags in VGPRs.
// ---------------------------------------------------------------------------
__global__ __launch_bounds__(256)
void msg_kernel(const float* __restrict__ relemb,   // (B,N,N,KG)
                const int*   __restrict__ adj,      // (B,N,N)
                const float* __restrict__ aBuf,     // (B*N, D)
                const float* __restrict__ bBuf,     // (B*N, D)
                const float* __restrict__ WcEff,    // (KG, D)
                const float* __restrict__ cbias,    // (D)
                float* __restrict__ sBuf) {         // (B*N, D)
    int bi = blockIdx.x;                // (b*N + i)
    int bb = bi / N_;
    int t = threadIdx.x;
    int lane = t & 31, w = t >> 5;
    int l16 = lane & 15, hi = lane >> 4;

    __shared__ __align__(16) __bf16 shA[2][16 * ALD];
    __shared__ float sh_aplus[D_];
    __shared__ float sh_adj[N_];
    sh_aplus[t] = aBuf[(size_t)bi * D_ + t] + cbias[t];
    sh_adj[t]   = (float)adj[(size_t)bi * N_ + t];

    // Cache this wave's two 16-col strips of WcEff as B fragments (VGPRs)
    v16bf bfrag[2][4];
#pragma unroll
    for (int dti = 0; dti < 2; ++dti)
#pragma unroll
        for (int kc = 0; kc < 4; ++kc)
            bfrag[dti][kc] = load_frag_B(WcEff, D_, kc * 32,
                                         (w * 2 + dti) * 16, lane);

    const float* Abase = relemb + (size_t)bi * N_ * KG_;
    float sacc0 = 0.f, sacc1 = 0.f;

    stage_tile(Abase, 0, shA[0], t);

    for (int jt = 0; jt < 16; ++jt) {
        __syncthreads();                       // staged tile jt ready
        if (jt + 1 < 16)
            stage_tile(Abase, jt + 1, shA[(jt + 1) & 1], t);

        const __bf16* tile = shA[jt & 1];
        v16bf af[4];
#pragma unroll
        for (int kc = 0; kc < 4; ++kc)
            af[kc] = frag_A_lds(tile, kc, lane);

#pragma unroll
        for (int dti = 0; dti < 2; ++dti) {
            v8f acc = {};
#pragma unroll
            for (int kc = 0; kc < 4; ++kc)
                acc = __builtin_amdgcn_wmma_f32_16x16x32_bf16(
                    false, af[kc], false, bfrag[dti][kc],
                    (short)0, acc, false, false);

            int col = (w * 2 + dti) * 16 + l16;
            float apc = sh_aplus[col];
            float part = 0.f;
#pragma unroll
            for (int r = 0; r < 8; ++r) {
                int j = jt * 16 + hi * 8 + r;
                float pre = acc[r] + apc +
                            bBuf[((size_t)bb * N_ + j) * D_ + col];
                pre = fmaxf(pre, 0.f);
                part += pre * sh_adj[j];
            }
            if (dti == 0) sacc0 += part; else sacc1 += part;
        }
    }

    // combine half-wave partials (lane L and L+16 hold same column)
    sacc0 += __shfl_xor(sacc0, 16, 32);
    sacc1 += __shfl_xor(sacc1, 16, 32);
    if (hi == 0) {
        sBuf[(size_t)bi * D_ + (w * 2 + 0) * 16 + l16] = sacc0;
        sBuf[(size_t)bi * D_ + (w * 2 + 1) * 16 + l16] = sacc1;
    }
}

// ---------------------------------------------------------------------------
// hout = LayerNorm(hin + tmp + deg*b2) ; one block per row (b*N+i)
// ---------------------------------------------------------------------------
__global__ __launch_bounds__(256)
void ln_kernel(const float* __restrict__ hin, const float* __restrict__ tmp,
               const int* __restrict__ adj, const float* __restrict__ b2l,
               const float* __restrict__ gamma, const float* __restrict__ beta,
               float* __restrict__ hout) {
    int row = blockIdx.x, t = threadIdx.x;
    __shared__ float red[D_];

    red[t] = (float)adj[(size_t)row * N_ + t];
    __syncthreads();
    for (int s = 128; s > 0; s >>= 1) {
        if (t < s) red[t] += red[t + s];
        __syncthreads();
    }
    float deg = red[0];
    __syncthreads();

    float x = hin[(size_t)row * D_ + t] + tmp[(size_t)row * D_ + t] +
              deg * b2l[t];
    red[t] = x;
    __syncthreads();
    for (int s = 128; s > 0; s >>= 1) {
        if (t < s) red[t] += red[t + s];
        __syncthreads();
    }
    float mu = red[0] * (1.f / D_);
    __syncthreads();

    float dx = x - mu;
    red[t] = dx * dx;
    __syncthreads();
    for (int s = 128; s > 0; s >>= 1) {
        if (t < s) red[t] += red[t + s];
        __syncthreads();
    }
    float var = red[0] * (1.f / D_);

    hout[(size_t)row * D_ + t] = dx * rsqrtf(var + 1e-5f) * gamma[t] + beta[t];
}

// ---------------------------------------------------------------------------
extern "C" void kernel_launch(void* const* d_in, const int* in_sizes, int n_in,
                              void* d_out, int out_size, void* d_ws, size_t ws_size,
                              hipStream_t stream) {
    const float* node_feats = (const float*)d_in[0];
    const float* relemb     = (const float*)d_in[1];
    const int*   adj        = (const int*)  d_in[2];
    const float* W_kg       = (const float*)d_in[3];
    const float* b_kg       = (const float*)d_in[4];
    const float* W1         = (const float*)d_in[5];
    const float* b1         = (const float*)d_in[6];
    const float* W2         = (const float*)d_in[7];
    const float* b2         = (const float*)d_in[8];
    const float* gamma      = (const float*)d_in[9];
    const float* beta       = (const float*)d_in[10];
    float* out = (float*)d_out;

    const int ROWS = B_ * N_;              // 1024
    float* ws     = (float*)d_ws;
    float* WcEff  = ws;                    // 2 * KG * D
    float* cbias  = WcEff + 2 * KG_ * D_;  // 2 * D
    float* aBuf   = cbias + 2 * D_;        // ROWS * D
    float* bBuf   = aBuf  + ROWS * D_;
    float* sBuf   = bBuf  + ROWS * D_;
    float* tmpBuf = sBuf  + ROWS * D_;
    float* h1     = tmpBuf + ROWS * D_;

    // Precompute WcEff[l] = W_kg @ Wc[l]  (128x256 -> 128 tiles -> 16 blocks)
    for (int l = 0; l < L_; ++l) {
        const float* Wc = W1 + ((size_t)l * 3 * D_ + 2 * D_) * D_;
        gemm_wmma<<<16, 256, 0, stream>>>(W_kg, Wc, WcEff + (size_t)l * KG_ * D_,
                                          KG_, D_, D_);
    }
    cbias_kernel<<<L_, D_, 0, stream>>>(b_kg, W1, b1, cbias);

    const float* hin = node_feats;
    for (int l = 0; l < L_; ++l) {
        float* hout = (l == L_ - 1) ? out : h1;
        const float* Wa = W1 + (size_t)l * 3 * D_ * D_;
        const float* Wb = Wa + (size_t)D_ * D_;

        // a = h @ Wa ; bfeat = h @ Wb   (1024 tiles -> 128 blocks each)
        gemm_wmma<<<128, 256, 0, stream>>>(hin, Wa, aBuf, ROWS, D_, D_);
        gemm_wmma<<<128, 256, 0, stream>>>(hin, Wb, bBuf, ROWS, D_, D_);

        // fused rel_emb GEMM + relu + adjacency-masked j-reduction
        msg_kernel<<<ROWS, 256, 0, stream>>>(relemb, adj, aBuf, bBuf,
                                             WcEff + (size_t)l * KG_ * D_,
                                             cbias + (size_t)l * D_, sBuf);

        // tmp = s @ W2
        gemm_wmma<<<128, 256, 0, stream>>>(sBuf, W2 + (size_t)l * D_ * D_,
                                           tmpBuf, ROWS, D_, D_);

        // h = LN(h + tmp + deg*b2)
        ln_kernel<<<ROWS, 256, 0, stream>>>(hin, tmpBuf, adj,
                                            b2 + (size_t)l * D_,
                                            gamma, beta, hout);
        hin = hout;
    }
}
